// BiMamba3Block_89610197663820
// MI455X (gfx1250) — matrix operand order repaired and checked by reading, not compiled, not run on hardware
//
#include <hip/hip_runtime.h>

// ---------------------------------------------------------------------------
// BiMamba3 block for MI455X (gfx1250, wave32, WMMA).
// All large GEMMs use v_wmma_f32_16x16x32_bf16 (bf16 in, f32 accumulate),
// register-blocked 32x64 per wave (8 WMMA tiles), 128x128 per workgroup.
// ---------------------------------------------------------------------------

typedef __attribute__((ext_vector_type(16))) __bf16 v16bf;
typedef __attribute__((ext_vector_type(8)))  float  v8f;

constexpr int BB  = 2;            // batch
constexpr int LL  = 2048;         // sequence length
constexpr int DM  = 1024;         // d_model (== COND)
constexpr int DI  = 2048;         // d_inner
constexpr int DS  = 64;           // d_state
constexpr int MT  = BB * LL;      // tokens = 4096
constexpr int MOD = 3 * DM;       // 3072
constexpr int PJ  = DI + 2 * DS;  // 2176

static __device__ __forceinline__ float siluf(float v) {
  return v / (1.f + __expf(-v));
}
static __device__ __forceinline__ float softplusf(float v) {
  return v > 20.f ? v : log1pf(__expf(v));
}

// ---- weight convert + transpose: Wt[n*K + k] = bf16(W[k*N + n]) ------------
__global__ void k_cvt_transpose(const float* __restrict__ W, __bf16* __restrict__ Wt,
                                int K, int N) {
  int n = blockIdx.x * 16 + threadIdx.x;
  int k = blockIdx.y * 16 + threadIdx.y;
  if (n < N && k < K) Wt[(size_t)n * K + k] = (__bf16)W[(size_t)k * N + n];
}

// ---- adaLN modulation: mod[b] = c[b] @ W + bias  (COND==DM) ----------------
__global__ void k_mod_gemm(const float* __restrict__ c, const float* __restrict__ w,
                           const float* __restrict__ bias, float* __restrict__ mod) {
  int n = blockIdx.x * blockDim.x + threadIdx.x;
  int b = blockIdx.y;
  if (n >= MOD) return;
  float acc = bias[n];
  const float* cr = c + (size_t)b * DM;
  for (int k = 0; k < DM; ++k) acc = fmaf(cr[k], w[(size_t)k * MOD + n], acc);
  mod[(size_t)b * MOD + n] = acc;
}

// ---- adaLN layernorm -> bf16 activations -----------------------------------
__global__ __launch_bounds__(256) void k_adaln_norm(const float* __restrict__ x,
                                                    const float* __restrict__ mod,
                                                    __bf16* __restrict__ h) {
  __shared__ float s1[256], s2[256];
  int tok = blockIdx.x;
  int b   = tok / LL;
  int tid = threadIdx.x;
  const float* xr = x + (size_t)tok * DM;
  float a1 = 0.f, a2 = 0.f;
  for (int i = tid; i < DM; i += 256) { float v = xr[i]; a1 += v; a2 += v * v; }
  s1[tid] = a1; s2[tid] = a2; __syncthreads();
  for (int off = 128; off > 0; off >>= 1) {
    if (tid < off) { s1[tid] += s1[tid + off]; s2[tid] += s2[tid + off]; }
    __syncthreads();
  }
  float mu  = s1[0] * (1.f / DM);
  float var = s2[0] * (1.f / DM) - mu * mu;
  float rs  = rsqrtf(var + 1e-5f);
  const float* mb = mod + (size_t)b * MOD;
  __bf16* hr = h + (size_t)tok * DM;
  for (int i = tid; i < DM; i += 256) {
    float v = (xr[i] - mu) * rs * (1.f + mb[DM + i]) + mb[i];
    hr[i] = (__bf16)v;
  }
}

// ---- WMMA bf16 GEMM: C[M,N] = A[M,K] @ Wt[N,K]^T ---------------------------
// Block = 8 waves (4 x 2); each wave owns a 32x64 tile = 2x4 WMMA tiles.
// Fragment layouts follow CDNA5 ISA §7.12.2 (wave32):
//  A 16x32 bf16: lanes 0-15 K 0..7/16..23, lanes 16-31 K 8..15/24..31
//  B 32x16 bf16: lanes 0-15 K 0..7/16..23, lanes 16-31 K 8..15/24..31 (2 per VGPR)
//  C/D: VGPR j -> M = j + 8*(lane>=16), N = lane&15
__global__ __launch_bounds__(256) void k_gemm_bf16(const __bf16* __restrict__ A,
                                                   const __bf16* __restrict__ Wt,
                                                   float* __restrict__ C,
                                                   int M, int N, int K) {
  int lane = threadIdx.x & 31;
  int w    = threadIdx.x >> 5;
  int wm   = w >> 1;  // 0..3
  int wn   = w & 1;   // 0..1
  int tm   = blockIdx.x * 128 + wm * 32;
  int tn   = blockIdx.y * 128 + wn * 64;

  v8f acc[2][4] = {};

  int mrow = tm + (lane & 15);
  int ncol = tn + (lane & 15);
  int ka   = (lane & 16) ? 8 : 0;
  int kb   = (lane & 16) ? 8 : 0;
  const __bf16* arow[2];
  arow[0] = A + (size_t)mrow * K + ka;
  arow[1] = arow[0] + (size_t)16 * K;
  const __bf16* brow[4];
#pragma unroll
  for (int j = 0; j < 4; ++j) brow[j] = Wt + (size_t)(ncol + 16 * j) * K + kb;

  for (int k0 = 0; k0 < K; k0 += 32) {
    v16bf fa[2], fb[4];
#pragma unroll
    for (int mi = 0; mi < 2; ++mi) {
      const __bf16* pa = arow[mi] + k0;
#pragma unroll
      for (int i = 0; i < 8; ++i) fa[mi][i] = pa[i];
#pragma unroll
      for (int i = 0; i < 8; ++i) fa[mi][8 + i] = pa[16 + i];
    }
#pragma unroll
    for (int nj = 0; nj < 4; ++nj) {
      const __bf16* pb = brow[nj] + k0;
#pragma unroll
      for (int i = 0; i < 8; ++i) fb[nj][i] = pb[i];
#pragma unroll
      for (int i = 0; i < 8; ++i) fb[nj][8 + i] = pb[16 + i];
    }
    // prefetch next K panel into caches (global_prefetch_b8)
    if (k0 + 64 < K) {
      __builtin_prefetch(arow[0] + k0 + 64, 0, 3);
      __builtin_prefetch(arow[1] + k0 + 64, 0, 3);
      __builtin_prefetch(brow[0] + k0 + 64, 0, 3);
      __builtin_prefetch(brow[2] + k0 + 64, 0, 3);
    }
#pragma unroll
    for (int mi = 0; mi < 2; ++mi)
#pragma unroll
      for (int nj = 0; nj < 4; ++nj)
        acc[mi][nj] = __builtin_amdgcn_wmma_f32_16x16x32_bf16(
            false, fa[mi], false, fb[nj], (short)0, acc[mi][nj], false, false);
  }

#pragma unroll
  for (int mi = 0; mi < 2; ++mi) {
    int cm = tm + 16 * mi + ((lane & 16) ? 8 : 0);
#pragma unroll
    for (int nj = 0; nj < 4; ++nj) {
      int cn = tn + 16 * nj + (lane & 15);
#pragma unroll
      for (int j = 0; j < 8; ++j) C[(size_t)(cm + j) * N + cn] = acc[mi][nj][j];
    }
  }
}

// ---- xz split: xs = silu(xz[:, :DI]) (f32 + bf16), z = xz[:, DI:] ----------
__global__ void k_silu_split(const float* __restrict__ xz, float* __restrict__ xs_f,
                             __bf16* __restrict__ xs_b, float* __restrict__ z) {
  size_t i = (size_t)blockIdx.x * blockDim.x + threadIdx.x;
  if (i >= (size_t)MT * DI) return;
  size_t tok = i / DI, j = i % DI;
  float xv = xz[tok * (2 * DI) + j];
  float zv = xz[tok * (2 * DI) + DI + j];
  float s  = siluf(xv);
  xs_f[i] = s;
  xs_b[i] = (__bf16)s;
  z[i]    = zv;
}

// ---- proj split: dt = softplus(+bias), Bm, Cm ------------------------------
__global__ void k_dtBC(const float* __restrict__ proj, const float* __restrict__ dt_bias,
                       float* __restrict__ dt, float* __restrict__ Bm,
                       float* __restrict__ Cm) {
  size_t i = (size_t)blockIdx.x * blockDim.x + threadIdx.x;
  if (i >= (size_t)MT * PJ) return;
  size_t tok = i / PJ;
  int    j   = (int)(i % PJ);
  float  v   = proj[i];
  if (j < DI)            dt[tok * DI + j]        = softplusf(v + dt_bias[j]);
  else if (j < DI + DS)  Bm[tok * DS + (j - DI)] = v;
  else                   Cm[tok * DS + (j - DI - DS)] = v;
}

// ---- A = -exp(A_log) -------------------------------------------------------
__global__ void k_prepA(const float* __restrict__ A_log, float* __restrict__ Af) {
  size_t i = (size_t)blockIdx.x * blockDim.x + threadIdx.x;
  if (i < (size_t)DI * DS) Af[i] = -__expf(A_log[i]);
}

// ---- selective scan (register-resident 64-wide state per channel) ----------
// y[t,d] = scan + x*D;  ybuf = bf16(y * silu(z))
__global__ __launch_bounds__(128) void k_scan(const float* __restrict__ dt,
                                              const float* __restrict__ Bm,
                                              const float* __restrict__ Cm,
                                              const float* __restrict__ xs,
                                              const float* __restrict__ z,
                                              const float* __restrict__ Af,
                                              const float* __restrict__ Dv,
                                              __bf16* __restrict__ ybuf,
                                              int reverse) {
  __shared__ float sB[DS], sC[DS];
  int b = blockIdx.y;
  int d = blockIdx.x * 128 + threadIdx.x;

  float st[DS];
#pragma unroll
  for (int s = 0; s < DS; ++s) st[s] = 0.f;
  float Av[DS];
#pragma unroll
  for (int s = 0; s < DS; ++s) Av[s] = Af[(size_t)d * DS + s];
  float Dd = Dv[d];

  for (int i = 0; i < LL; ++i) {
    int t = reverse ? (LL - 1 - i) : i;
    size_t tok = (size_t)b * LL + t;
    __syncthreads();
    if (threadIdx.x < DS)            sB[threadIdx.x]      = Bm[tok * DS + threadIdx.x];
    else if (threadIdx.x < 2 * DS)   sC[threadIdx.x - DS] = Cm[tok * DS + threadIdx.x - DS];
    __syncthreads();

    float dtt = dt[tok * DI + d];
    float xt  = xs[tok * DI + d];
    float dx  = dtt * xt;
    float y   = 0.f;
#pragma unroll
    for (int s = 0; s < DS; ++s) {
      float dA = __expf(dtt * Av[s]);
      st[s] = st[s] * dA + dx * sB[s];
      y += st[s] * sC[s];
    }
    y += xt * Dd;
    float zt = z[tok * DI + d];
    ybuf[tok * DI + d] = (__bf16)(y * siluf(zt));
  }
}

// ---- x2 = x + gate_m * (out_f + out_b) -------------------------------------
__global__ void k_combine(const float* __restrict__ x, const float* __restrict__ mod,
                          const float* __restrict__ of, const float* __restrict__ ob,
                          float* __restrict__ x2) {
  size_t i = (size_t)blockIdx.x * blockDim.x + threadIdx.x;
  if (i >= (size_t)MT * DM) return;
  size_t tok = i / DM;
  int    ch  = (int)(i % DM);
  int    b   = (int)(tok / LL);
  float  g   = mod[(size_t)b * MOD + 2 * DM + ch];
  x2[i] = x[i] + g * (of[i] + ob[i]);
}

// ---- m = bf16(silu(u1) * u2) -----------------------------------------------
__global__ void k_glu(const float* __restrict__ u1, const float* __restrict__ u2,
                      __bf16* __restrict__ m) {
  size_t i = (size_t)blockIdx.x * blockDim.x + threadIdx.x;
  if (i >= (size_t)MT * DI) return;
  m[i] = (__bf16)(siluf(u1[i]) * u2[i]);
}

// ---- out = x2 + gate_p * mlp_out -------------------------------------------
__global__ void k_final(const float* __restrict__ x2, const float* __restrict__ mod,
                        const float* __restrict__ mlp, float* __restrict__ out) {
  size_t i = (size_t)blockIdx.x * blockDim.x + threadIdx.x;
  if (i >= (size_t)MT * DM) return;
  size_t tok = i / DM;
  int    ch  = (int)(i % DM);
  int    b   = (int)(tok / LL);
  float  g   = mod[(size_t)b * MOD + 2 * DM + ch];
  out[i] = x2[i] + g * mlp[i];
}

// ===========================================================================
extern "C" void kernel_launch(void* const* d_in, const int* in_sizes, int n_in,
                              void* d_out, int out_size, void* d_ws, size_t ws_size,
                              hipStream_t stream) {
  (void)in_sizes; (void)n_in; (void)out_size; (void)ws_size;

  const float* x            = (const float*)d_in[0];
  const float* c            = (const float*)d_in[1];
  const float* adaln_m_w    = (const float*)d_in[2];
  const float* adaln_m_b    = (const float*)d_in[3];
  const float* adaln_p_w    = (const float*)d_in[4];
  const float* adaln_p_b    = (const float*)d_in[5];
  const float* mlp_w1       = (const float*)d_in[6];
  const float* mlp_w2       = (const float*)d_in[7];
  const float* mlp_w3       = (const float*)d_in[8];
  const float* in_w[2]      = {(const float*)d_in[9],  (const float*)d_in[15]};
  const float* xproj_w[2]   = {(const float*)d_in[10], (const float*)d_in[16]};
  const float* dt_bias[2]   = {(const float*)d_in[11], (const float*)d_in[17]};
  const float* A_log[2]     = {(const float*)d_in[12], (const float*)d_in[18]};
  const float* Dv[2]        = {(const float*)d_in[13], (const float*)d_in[19]};
  const float* out_w[2]     = {(const float*)d_in[14], (const float*)d_in[20]};
  float* out = (float*)d_out;

  // ---- workspace bump allocator -------------------------------------------
  char*  ws  = (char*)d_ws;
  size_t off = 0;
  auto alloc = [&](size_t bytes) -> void* {
    void* p = ws + off;
    off = (off + bytes + 255) & ~(size_t)255;
    return p;
  };

  // bf16 transposed weights (persistent across phases)
  __bf16* in_w_t[2]   = {(__bf16*)alloc((size_t)2*DI*DM*2),  (__bf16*)alloc((size_t)2*DI*DM*2)};
  __bf16* xproj_t[2]  = {(__bf16*)alloc((size_t)PJ*DI*2),    (__bf16*)alloc((size_t)PJ*DI*2)};
  __bf16* out_w_t[2]  = {(__bf16*)alloc((size_t)DM*DI*2),    (__bf16*)alloc((size_t)DM*DI*2)};
  __bf16* w1t         = (__bf16*)alloc((size_t)DI*DM*2);
  __bf16* w2t         = (__bf16*)alloc((size_t)DI*DM*2);
  __bf16* w3t         = (__bf16*)alloc((size_t)DM*DI*2);
  float*  Af[2]       = {(float*)alloc((size_t)DI*DS*4),     (float*)alloc((size_t)DI*DS*4)};

  // activations
  float*  modm   = (float*)alloc((size_t)BB*MOD*4);
  float*  modp   = (float*)alloc((size_t)BB*MOD*4);
  __bf16* h_b    = (__bf16*)alloc((size_t)MT*DM*2);    // reused for norm2 output
  float*  xz     = (float*)alloc((size_t)MT*2*DI*4);   // reused: u1 (first half), u2 (second half)
  float*  xs_f   = (float*)alloc((size_t)MT*DI*4);
  __bf16* xs_b   = (__bf16*)alloc((size_t)MT*DI*2);    // reused for GLU output m
  float*  zbuf   = (float*)alloc((size_t)MT*DI*4);
  float*  proj   = (float*)alloc((size_t)MT*PJ*4);     // reused for mlp_out
  float*  dtbuf  = (float*)alloc((size_t)MT*DI*4);
  float*  Bm     = (float*)alloc((size_t)MT*DS*4);
  float*  Cm     = (float*)alloc((size_t)MT*DS*4);
  __bf16* ybuf   = (__bf16*)alloc((size_t)MT*DI*2);
  float*  dirout[2] = {(float*)alloc((size_t)MT*DM*4), (float*)alloc((size_t)MT*DM*4)};
  float*  x2     = (float*)alloc((size_t)MT*DM*4);
  float*  u1     = xz;
  float*  u2     = xz + (size_t)MT*DI;
  float*  mlpout = proj;

  dim3 b16(16, 16);
  auto tgrid = [](int N, int K) { return dim3((N + 15) / 16, (K + 15) / 16); };

  // ---- 1. weight conversion ------------------------------------------------
  for (int d = 0; d < 2; ++d) {
    k_cvt_transpose<<<tgrid(2*DI, DM), b16, 0, stream>>>(in_w[d],    in_w_t[d],  DM, 2*DI);
    k_cvt_transpose<<<tgrid(PJ,  DI),  b16, 0, stream>>>(xproj_w[d], xproj_t[d], DI, PJ);
    k_cvt_transpose<<<tgrid(DM,  DI),  b16, 0, stream>>>(out_w[d],   out_w_t[d], DI, DM);
    k_prepA<<<(DI*DS + 255)/256, 256, 0, stream>>>(A_log[d], Af[d]);
  }
  k_cvt_transpose<<<tgrid(DI, DM), b16, 0, stream>>>(mlp_w1, w1t, DM, DI);
  k_cvt_transpose<<<tgrid(DI, DM), b16, 0, stream>>>(mlp_w2, w2t, DM, DI);
  k_cvt_transpose<<<tgrid(DM, DI), b16, 0, stream>>>(mlp_w3, w3t, DI, DM);

  // ---- 2. adaLN modulations -----------------------------------------------
  k_mod_gemm<<<dim3((MOD + 255)/256, BB), 256, 0, stream>>>(c, adaln_m_w, adaln_m_b, modm);
  k_mod_gemm<<<dim3((MOD + 255)/256, BB), 256, 0, stream>>>(c, adaln_p_w, adaln_p_b, modp);

  // ---- 3. norm 1 -----------------------------------------------------------
  k_adaln_norm<<<MT, 256, 0, stream>>>(x, modm, h_b);

  // ---- 4. bidirectional SSM (flip == reversed-time scan) -------------------
  const size_t nDI = (size_t)MT * DI, nPJ = (size_t)MT * PJ, nDM = (size_t)MT * DM;
  for (int d = 0; d < 2; ++d) {
    k_gemm_bf16<<<dim3(MT/128, (2*DI)/128), 256, 0, stream>>>(h_b, in_w_t[d], xz, MT, 2*DI, DM);
    k_silu_split<<<(nDI + 255)/256, 256, 0, stream>>>(xz, xs_f, xs_b, zbuf);
    k_gemm_bf16<<<dim3(MT/128, PJ/128), 256, 0, stream>>>(xs_b, xproj_t[d], proj, MT, PJ, DI);
    k_dtBC<<<(nPJ + 255)/256, 256, 0, stream>>>(proj, dt_bias[d], dtbuf, Bm, Cm);
    k_scan<<<dim3(DI/128, BB), 128, 0, stream>>>(dtbuf, Bm, Cm, xs_f, zbuf,
                                                 Af[d], Dv[d], ybuf, /*reverse=*/d);
    k_gemm_bf16<<<dim3(MT/128, DM/128), 256, 0, stream>>>(ybuf, out_w_t[d], dirout[d], MT, DM, DI);
  }
  k_combine<<<(nDM + 255)/256, 256, 0, stream>>>(x, modm, dirout[0], dirout[1], x2);

  // ---- 5. MLP branch -------------------------------------------------------
  k_adaln_norm<<<MT, 256, 0, stream>>>(x2, modp, h_b);
  k_gemm_bf16<<<dim3(MT/128, DI/128), 256, 0, stream>>>(h_b, w1t, u1, MT, DI, DM);
  k_gemm_bf16<<<dim3(MT/128, DI/128), 256, 0, stream>>>(h_b, w2t, u2, MT, DI, DM);
  k_glu<<<(nDI + 255)/256, 256, 0, stream>>>(u1, u2, xs_b);
  k_gemm_bf16<<<dim3(MT/128, DM/128), 256, 0, stream>>>(xs_b, w3t, mlpout, MT, DM, DI);
  k_final<<<(nDM + 255)/256, 256, 0, stream>>>(x2, modp, mlpout, out);
}
